// InterpretableMultiHeadAttention_49108656063270
// MI455X (gfx1250) — matrix operand order repaired and checked by reading, not compiled, not run on hardware
//
#include <hip/hip_runtime.h>
#include <hip/hip_bf16.h>

// ---------------------------------------------------------------------------
// MI455X (gfx1250) multi-head attention forward.
//  - All matmuls: v_wmma_f32_16x16x32_bf16 (wave32, f32 accumulate).
//  - GEMMs: branchless peeled register pipeline (loads for k are issued while
//    WMMAs for k-32 execute; last iteration peeled so no guard/copies exist
//    for the scheduler to collapse).
//  - Attention: K tiles streamed by the Tensor Data Mover through a 4-deep
//    LDS pipeline (s_wait_tensorcnt 3 steady state), V^T tiles double
//    buffered; scores staged in LDS (16x2048 f32 = 128 KB of the 320 KB WGP
//    LDS); the 512 MB attn_weights tensor is written to HBM exactly once.
// ---------------------------------------------------------------------------

typedef __bf16 bf16;
typedef __attribute__((ext_vector_type(16))) __bf16 v16bf;
typedef __attribute__((ext_vector_type(8)))  __bf16 v8bf;
typedef __attribute__((ext_vector_type(8)))  float  v8f;
typedef __attribute__((ext_vector_type(4)))  float  v4f;
typedef __attribute__((ext_vector_type(4)))  unsigned int u32x4;
typedef __attribute__((ext_vector_type(8)))  int  i32x8;
typedef __attribute__((ext_vector_type(4)))  int  i32x4;

constexpr int Bc  = 2;
constexpr int Hc  = 16;
constexpr int Sc  = 2048;
constexpr int Dc  = 1024;
constexpr int HDc = 64;
constexpr int Mrows = Bc * Sc;   // 4096

#ifndef __has_builtin
#define __has_builtin(x) 0
#endif
#if __has_builtin(__builtin_amdgcn_tensor_load_to_lds) && \
    __has_builtin(__builtin_amdgcn_s_wait_tensorcnt)
#define USE_TDM 1
#else
#define USE_TDM 0
#endif

__device__ __forceinline__ bf16 f2bf(float f) {
  union { float f; unsigned u; } v; v.f = f;
  unsigned r = v.u + 0x7FFFu + ((v.u >> 16) & 1u);   // round-to-nearest-even
  unsigned short h = (unsigned short)(r >> 16);
  return __builtin_bit_cast(bf16, h);
}

__device__ __forceinline__ v16bf cat16(v8bf lo, v8bf hi) {
  v16bf r;
#pragma unroll
  for (int i = 0; i < 8; ++i) { r[i] = lo[i]; r[i + 8] = hi[i]; }
  return r;
}

// A-matrix 16x32 bf16 fragment (ISA 7.12.2): lanes 0-15 hold M=lane,
// K = 0..7 / 16..23; lanes 16-31 hold K = 8..15 / 24..31.
// Row-major [M, ld] memory -> two contiguous 16-byte loads per lane.
__device__ __forceinline__ v16bf load_a_frag(const bf16* base, int ld, int m0, int k0) {
  int lane = threadIdx.x & 31;
  int m    = lane & 15;
  int cb   = (lane < 16) ? 0 : 8;
  const bf16* p = base + (size_t)(m0 + m) * ld + k0 + cb;
  return cat16(*(const v8bf*)p, *(const v8bf*)(p + 16));
}

// B-matrix 32x16 bf16 fragment: lane n holds column n; lanes 0-15 K=0..7/16..23,
// lanes 16-31 K=8..15/24..31.  B[k,n] = Mem[(n0+n)*ld + k0 + k] (contraction dim
// contiguous) -> two contiguous b128 loads.  Works for global or LDS pointers.
__device__ __forceinline__ v16bf load_b_frag(const bf16* base, int ld, int n0, int k0) {
  int lane = threadIdx.x & 31;
  int n    = lane & 15;
  int kb   = (lane < 16) ? 0 : 8;
  const bf16* p = base + (size_t)(n0 + n) * ld + k0 + kb;
  return cat16(*(const v8bf*)p, *(const v8bf*)(p + 16));
}

__device__ __forceinline__ v8f wmma_bf16(v16bf a, v16bf b, v8f c) {
  return __builtin_amdgcn_wmma_f32_16x16x32_bf16(
      false, a, false, b, (short)0, c, false, false);
}

#if USE_TDM
// 2-D TENSOR_LOAD_TO_LDS.  D# per CDNA5 ISA ch.8:
//   group0: [1:0]=count(1) | lds_addr[63:32] | global_addr[120:64] | type=2
//   group1: data_size=1(2B), tensor_dim0/1, tile_dim0/1, tensor_dim0_stride.
// Dims/strides in elements; lds_off in bytes; groups 2/3 zero (<=2D tensor).
__device__ __forceinline__ void tdm_load_2d(const void* gptr, unsigned lds_off,
                                            unsigned tile_d0, unsigned tile_d1,
                                            unsigned tens_d0, unsigned tens_d1,
                                            unsigned stride0) {
  unsigned long long ga = (unsigned long long)(uintptr_t)gptr;
  u32x4 g0;
  g0[0] = 1u;                                  // count=1, user descriptor
  g0[1] = lds_off;                             // lds_addr (bytes)
  g0[2] = (unsigned)ga;                        // global_addr[31:0]
  g0[3] = (unsigned)((ga >> 32) & 0x01FFFFFFu) | (2u << 30);   // [120:64], type=2
  i32x8 g1;
  g1[0] = (int)(1u << 16);                     // data_size=1 -> 2 bytes/elem
  g1[1] = (int)((tens_d0 & 0xFFFFu) << 16);                    // tensor_dim0 lo
  g1[2] = (int)(((tens_d0 >> 16) & 0xFFFFu) | ((tens_d1 & 0xFFFFu) << 16));
  g1[3] = (int)(((tens_d1 >> 16) & 0xFFFFu) | ((tile_d0 & 0xFFFFu) << 16));
  g1[4] = (int)(tile_d1 & 0xFFFFu);            // tile_dim1 (tile_dim2 = 0)
  g1[5] = (int)stride0;                        // tensor_dim0_stride[31:0]
  g1[6] = 0;
  g1[7] = 0;
  i32x4 z4 = {0, 0, 0, 0};
#if __clang_major__ >= 23
  i32x8 z8 = {0, 0, 0, 0, 0, 0, 0, 0};
  __builtin_amdgcn_tensor_load_to_lds(g0, g1, z4, z4, z8, 0);
#else
  __builtin_amdgcn_tensor_load_to_lds(g0, g1, z4, z4, 0);
#endif
}
#endif  // USE_TDM

// ---------------------------------------------------------------------------
// f32 -> bf16 conversion
// ---------------------------------------------------------------------------
__global__ void cvt_bf16_kernel(const float* __restrict__ src,
                                bf16* __restrict__ dst, int n) {
  int i = (blockIdx.x * blockDim.x + threadIdx.x) * 4;
  if (i + 3 < n) {
    v4f x = *(const v4f*)(src + i);
    dst[i + 0] = f2bf(x[0]); dst[i + 1] = f2bf(x[1]);
    dst[i + 2] = f2bf(x[2]); dst[i + 3] = f2bf(x[3]);
  }
}

// ---------------------------------------------------------------------------
// GEMM: C[m,e] = sum_d X[m,d] * W[e,d] + bias[e]     (torch Linear)
// Each wave computes a 16x64 strip (4 n-tiles reusing one A fragment).
// Branchless software pipeline: the loop body loads fragments for kk and
// multiplies fragments for kk-32; the final multiply is peeled out.
// mode 0: bf16 out, head-split [B,H,S,hd];  mode 1: bf16 out, V^T [B,H,hd,S];
// mode 2: f32 out, flat [M, D].
// ---------------------------------------------------------------------------
__global__ __launch_bounds__(128) void gemm_kernel(
    const bf16* __restrict__ X, const bf16* __restrict__ W,
    const float* __restrict__ bias, int mode,
    bf16* __restrict__ outb, float* __restrict__ outf) {
  const int wave = threadIdx.x >> 5;
  const int lane = threadIdx.x & 31;
  const int m0 = blockIdx.x * 16;
  const int n0 = (blockIdx.y * 4 + wave) * 64;

  v8f c[4] = {};
  v16bf a  = load_a_frag(X, Dc, m0, 0);
  v16bf b0 = load_b_frag(W, Dc, n0,      0);
  v16bf b1 = load_b_frag(W, Dc, n0 + 16, 0);
  v16bf b2 = load_b_frag(W, Dc, n0 + 32, 0);
  v16bf b3 = load_b_frag(W, Dc, n0 + 48, 0);

  for (int kk = 32; kk < Dc; kk += 32) {
    v16bf an  = load_a_frag(X, Dc, m0, kk);
    v16bf b0n = load_b_frag(W, Dc, n0,      kk);
    v16bf b1n = load_b_frag(W, Dc, n0 + 16, kk);
    v16bf b2n = load_b_frag(W, Dc, n0 + 32, kk);
    v16bf b3n = load_b_frag(W, Dc, n0 + 48, kk);
    c[0] = wmma_bf16(a, b0, c[0]);
    c[1] = wmma_bf16(a, b1, c[1]);
    c[2] = wmma_bf16(a, b2, c[2]);
    c[3] = wmma_bf16(a, b3, c[3]);
    a = an; b0 = b0n; b1 = b1n; b2 = b2n; b3 = b3n;
  }
  c[0] = wmma_bf16(a, b0, c[0]);               // peeled epilogue
  c[1] = wmma_bf16(a, b1, c[1]);
  c[2] = wmma_bf16(a, b2, c[2]);
  c[3] = wmma_bf16(a, b3, c[3]);

  // C layout: lanes 0-15: M = r, N = lane; lanes 16-31: M = 8+r, N = lane-16.
  const int n  = lane & 15;
  const int mh = (lane < 16) ? 0 : 8;
#pragma unroll
  for (int i = 0; i < 4; ++i) {
    float bn = bias[n0 + i * 16 + n];
#pragma unroll
    for (int r = 0; r < 8; ++r) {
      float v = c[i][r] + bn;
      int m = m0 + mh + r;
      int e = n0 + i * 16 + n;
      if (mode == 2) {
        outf[(size_t)m * Dc + e] = v;
      } else {
        int bb = m >> 11, s = m & (Sc - 1);
        int h  = e >> 6,  hcol = e & (HDc - 1);
        if (mode == 0)
          outb[(((size_t)(bb * Hc + h)) * Sc + s) * HDc + hcol] = f2bf(v);
        else
          outb[(((size_t)(bb * Hc + h)) * HDc + hcol) * Sc + s] = f2bf(v);
      }
    }
  }
}

// ---------------------------------------------------------------------------
// Attention: one wave per (b, h, 16-row q-tile).
// LDS: [0, 128K)    scores/probs, 16 x 2048 f32
//      [128K,+8K)   K-tile quad buffer, 4 x (16 x 64) bf16   (TDM dest)
//      [136K,+8K)   V-tile double buffer, 2 x (64 x 32) bf16 (TDM dest)
// ---------------------------------------------------------------------------
constexpr unsigned KT_OFF = 16u * Sc * 4u;          // 131072
constexpr unsigned VT_OFF = KT_OFF + 8192u;         // 139264
constexpr unsigned SMEM_BYTES = VT_OFF + 8192u;     // 147456

__global__ __launch_bounds__(32) void attn_kernel(
    const bf16* __restrict__ Qh, const bf16* __restrict__ Kh,
    const bf16* __restrict__ Vt, float* __restrict__ attn,
    bf16* __restrict__ ctx) {
  extern __shared__ char smem[];
  float* sS = (float*)smem;

  const int bid = blockIdx.x;                   // (b*H + h) * 128 + t
  const int t   = bid & 127;
  const int bh  = bid >> 7;
  const int q0  = t * 16;
  const int lane = threadIdx.x;
  const int n   = lane & 15;
  const int mh  = (lane < 16) ? 0 : 8;

  const bf16* Qbase = Qh + (size_t)bh * Sc * HDc;
  const bf16* Kbase = Kh + (size_t)bh * Sc * HDc;
  const bf16* Vbase = Vt + (size_t)bh * HDc * Sc;

  // Q tile 16x64 stays resident in VGPRs for the whole kernel.
  v16bf aq0 = load_a_frag(Qbase, HDc, q0, 0);
  v16bf aq1 = load_a_frag(Qbase, HDc, q0, 32);

  // ---- Phase 1: scores -> LDS (4-deep TDM K-tile pipeline) ----
#if USE_TDM
  bf16* ktile = (bf16*)(smem + KT_OFF);
#pragma unroll
  for (int p = 0; p < 4; ++p)                   // prime tiles 0..min(3,t)
    if (p <= t)
      tdm_load_2d(Kbase + (size_t)p * 16 * HDc, KT_OFF + (unsigned)(p & 3) * 2048,
                  64, 16, 64, Sc, 64);
#endif
  for (int j = 0; j <= t; ++j) {
    int k0 = j * 16;
    v16bf b0, b1;
#if USE_TDM
    // Tile j is complete when tensorcnt <= (#tiles issued after j).
    int o = ((t < j + 3) ? t : (j + 3)) - j;
    if (o >= 3)      __builtin_amdgcn_s_wait_tensorcnt(3);
    else if (o == 2) __builtin_amdgcn_s_wait_tensorcnt(2);
    else if (o == 1) __builtin_amdgcn_s_wait_tensorcnt(1);
    else             __builtin_amdgcn_s_wait_tensorcnt(0);
    __asm__ volatile("" ::: "memory");          // keep ds reads below the wait
    const bf16* tile = ktile + (j & 3) * 1024;
    b0 = load_b_frag(tile, HDc, 0, 0);          // ds_load_b128 x2
    b1 = load_b_frag(tile, HDc, 0, 32);
#else
    b0 = load_b_frag(Kbase, HDc, k0, 0);
    b1 = load_b_frag(Kbase, HDc, k0, 32);
#endif
    v8f c = {};
    c = wmma_bf16(aq0, b0, c);
    c = wmma_bf16(aq1, b1, c);
#pragma unroll
    for (int r = 0; r < 8; ++r) {
      int mg = q0 + mh + r;
      int kg = k0 + n;
      float v = c[r] * 0.125f;                  // 1/sqrt(64)
      if (kg > mg) v = -__builtin_inff();       // causal mask
      sS[(size_t)(mh + r) * Sc + kg] = v;
    }
#if USE_TDM
    // Tile j consumed (wmma waited on its ds reads) -> its buffer is free.
    if (j + 4 <= t)
      tdm_load_2d(Kbase + (size_t)(k0 + 64) * HDc,
                  KT_OFF + (unsigned)((j + 4) & 3) * 2048, 64, 16, 64, Sc, 64);
#endif
  }

  // ---- Phase 2: softmax; write 512MB of probs to HBM exactly once ----
  float* attn_base = attn + ((size_t)bh * Sc + q0) * Sc;
  for (int r = 0; r < 16; ++r) {
    int q = q0 + r;
    float* row = sS + (size_t)r * Sc;
    float mx = -__builtin_inff();
    for (int c = lane; c <= q; c += 32) mx = fmaxf(mx, row[c]);
#pragma unroll
    for (int off = 16; off > 0; off >>= 1) mx = fmaxf(mx, __shfl_xor(mx, off, 32));
    float sum = 0.0f;
    for (int c = lane; c <= q; c += 32) {       // store exp once, reuse below
      float e = __expf(row[c] - mx);
      sum += e;
      row[c] = e;
    }
#pragma unroll
    for (int off = 16; off > 0; off >>= 1) sum += __shfl_xor(sum, off, 32);
    float inv = 1.0f / sum;
    float* arow = attn_base + (size_t)r * Sc;
    for (int c = lane; c < Sc; c += 32) {
      float p = (c <= q) ? row[c] * inv : 0.0f;
      arow[c] = p;
      row[c]  = p;                              // keep probs in LDS for P @ V
    }
  }

  // ---- Phase 3: context = P @ V ----
  v8f acc[4] = {};
  const int nchunks = (q0 + 16 + 31) >> 5;      // k-chunks of 32 up to q0+15
#if USE_TDM
  bf16* vtile = (bf16*)(smem + VT_OFF);
  // V^T tile: 64 rows (hd) x 32 cols (keys), row stride Sc.
  tdm_load_2d(Vbase, VT_OFF, 32, 64, Sc, 64, Sc);
#endif
  for (int j = 0; j < nchunks; ++j) {
    int k0 = j * 32;
    // P fragment from LDS probs (f32 -> bf16).
    int m  = lane & 15;
    int cb = (lane < 16) ? 0 : 8;
    const v4f* p4 = (const v4f*)(sS + (size_t)m * Sc + k0 + cb);
    v4f x0 = p4[0], x1 = p4[1], x2 = p4[4], x3 = p4[5];
    v16bf pa;
#pragma unroll
    for (int i = 0; i < 4; ++i) {
      pa[i]      = f2bf(x0[i]);
      pa[4 + i]  = f2bf(x1[i]);
      pa[8 + i]  = f2bf(x2[i]);
      pa[12 + i] = f2bf(x3[i]);
    }
#if USE_TDM
    if (j + 1 < nchunks) {
      tdm_load_2d(Vbase + (size_t)(k0 + 32),
                  VT_OFF + (unsigned)(((j + 1) & 1) * 4096),
                  32, 64, Sc, 64, Sc);
      __builtin_amdgcn_s_wait_tensorcnt(1);
    } else {
      __builtin_amdgcn_s_wait_tensorcnt(0);
    }
    __asm__ volatile("" ::: "memory");
    const bf16* vt = vtile + (j & 1) * 2048;    // 64x32 tile, ld=32
#pragma unroll
    for (int i = 0; i < 4; ++i) {
      v16bf bv = load_b_frag(vt, 32, i * 16, 0);
      acc[i] = wmma_bf16(pa, bv, acc[i]);
    }
#else
#pragma unroll
    for (int i = 0; i < 4; ++i) {
      v16bf bv = load_b_frag(Vbase, Sc, i * 16, k0);
      acc[i] = wmma_bf16(pa, bv, acc[i]);
    }
#endif
  }

  // ctx merged-head layout [B*S, D] (bf16) for the output projection.
  const int h  = bh & (Hc - 1);
  const int bb = bh >> 4;
#pragma unroll
  for (int i = 0; i < 4; ++i) {
#pragma unroll
    for (int r = 0; r < 8; ++r) {
      int row = bb * Sc + q0 + mh + r;
      int col = h * HDc + i * 16 + n;
      ctx[(size_t)row * Dc + col] = f2bf(acc[i][r]);
    }
  }
}

// ---------------------------------------------------------------------------
extern "C" void kernel_launch(void* const* d_in, const int* in_sizes, int n_in,
                              void* d_out, int out_size, void* d_ws, size_t ws_size,
                              hipStream_t stream) {
  const float* q  = (const float*)d_in[0];
  const float* k  = (const float*)d_in[1];
  const float* v  = (const float*)d_in[2];
  const float* Wq = (const float*)d_in[3];
  const float* bq = (const float*)d_in[4];
  const float* Wk = (const float*)d_in[5];
  const float* bk = (const float*)d_in[6];
  const float* Wv = (const float*)d_in[7];
  const float* bv = (const float*)d_in[8];
  const float* Wo = (const float*)d_in[9];
  const float* bo = (const float*)d_in[10];

  // Workspace layout (bf16 scratch, ~64 MB total).
  char* ws = (char*)d_ws;
  size_t off = 0;
  auto alloc = [&](size_t bytes) {
    char* p = ws + off;
    off += (bytes + 255) & ~(size_t)255;
    return p;
  };
  const size_t actN = (size_t)Mrows * Dc;       // 4M elements
  const size_t wN   = (size_t)Dc * Dc;          // 1M elements
  bf16* qbf  = (bf16*)alloc(actN * 2);
  bf16* kbf  = (bf16*)alloc(actN * 2);
  bf16* vbf  = (bf16*)alloc(actN * 2);
  bf16* Wqb  = (bf16*)alloc(wN * 2);
  bf16* Wkb  = (bf16*)alloc(wN * 2);
  bf16* Wvb  = (bf16*)alloc(wN * 2);
  bf16* Wob  = (bf16*)alloc(wN * 2);
  bf16* Qh   = (bf16*)alloc(actN * 2);          // [B,H,S,hd]
  bf16* Kh   = (bf16*)alloc(actN * 2);          // [B,H,S,hd]
  bf16* Vt   = (bf16*)alloc(actN * 2);          // [B,H,hd,S]
  bf16* ctx  = (bf16*)alloc(actN * 2);          // [B*S, D]

  auto cvt = [&](const float* src, bf16* dst, size_t n) {
    int blocks = (int)((n / 4 + 255) / 256);
    cvt_bf16_kernel<<<blocks, 256, 0, stream>>>(src, dst, (int)n);
  };
  cvt(q,  qbf, actN);
  cvt(k,  kbf, actN);
  cvt(v,  vbf, actN);
  cvt(Wq, Wqb, wN);
  cvt(Wk, Wkb, wN);
  cvt(Wv, Wvb, wN);
  cvt(Wo, Wob, wN);

  dim3 gb(Mrows / 16, Dc / 256);                // (256, 4): 4 waves x 16x64 strip
  dim3 tb(128);
  gemm_kernel<<<gb, tb, 0, stream>>>(qbf, Wqb, bq, 0, Qh, nullptr);
  gemm_kernel<<<gb, tb, 0, stream>>>(kbf, Wkb, bk, 0, Kh, nullptr);
  gemm_kernel<<<gb, tb, 0, stream>>>(vbf, Wvb, bv, 1, Vt, nullptr);

  float* out_proj = (float*)d_out;                       // [B,S,D]
  float* attn_w   = out_proj + (size_t)Mrows * Dc;       // [B,H,S,S]
  attn_kernel<<<Bc * Hc * (Sc / 16), 32, SMEM_BYTES, stream>>>(
      Qh, Kh, Vt, attn_w, ctx);

  gemm_kernel<<<gb, tb, 0, stream>>>(ctx, Wob, bo, 2, nullptr, out_proj);
}
